// CoarseMatching_6923487281717
// MI455X (gfx1250) — matrix operand order repaired and checked by reading, not compiled, not run on hardware
//
#include <hip/hip_runtime.h>

// Problem constants (from setup_inputs: n=4, h=w=64, c=256)
#define N_BATCH 4
#define L_DIM   4096
#define S_DIM   4096
#define C_DIM   256
#define HW      64
#define BORDER_R 2
#define THRESH  0.2f
// sim = dot * (1/c) / temperature = dot / (256 * 0.1)
#define SIM_SCALE (1.0f / 25.6f)

#define NE_INPUT ((size_t)N_BATCH * L_DIM * C_DIM)        // 4,194,304 elems per input
#define STATS_BYTES ((size_t)4 * N_BATCH * 4096 * 4)      // 256 KB

typedef __attribute__((ext_vector_type(16))) __bf16         v16bf;
typedef __attribute__((ext_vector_type(8)))  float          v8f;
typedef __attribute__((ext_vector_type(8)))  unsigned short ushort8;
typedef __attribute__((ext_vector_type(4)))  unsigned short ushort4v;

union FVec8 { v8f v; float f[8]; };

__device__ __forceinline__ unsigned short f32_bf16_rne(float x) {
  unsigned u = __float_as_uint(x);
  u += 0x7fffu + ((u >> 16) & 1u);
  return (unsigned short)(u >> 16);
}

// -------- one-shot split pass: f32 -> (hi bf16, lo bf16) residual pair --------
__global__ __launch_bounds__(256) void split_bf16_kernel(const float* __restrict__ src,
                                                         unsigned short* __restrict__ hi,
                                                         unsigned short* __restrict__ lo,
                                                         int n4) {
  const int i = blockIdx.x * 256 + threadIdx.x;
  if (i >= n4) return;
  const float4 f = reinterpret_cast<const float4*>(src)[i];
  const float v[4] = {f.x, f.y, f.z, f.w};
  unsigned short h[4], l[4];
#pragma unroll
  for (int e = 0; e < 4; ++e) {
    h[e] = f32_bf16_rne(v[e]);
    const float hf = __uint_as_float(((unsigned)h[e]) << 16);
    l[e] = f32_bf16_rne(v[e] - hf);
  }
  *reinterpret_cast<ushort4v*>(hi + (size_t)4 * i) = (ushort4v){h[0], h[1], h[2], h[3]};
  *reinterpret_cast<ushort4v*>(lo + (size_t)4 * i) = (ushort4v){l[0], l[1], l[2], l[3]};
}

// Load a v16bf WMMA fragment as two contiguous 16-byte chunks (no VALU work).
__device__ __forceinline__ v16bf load_frag_bf(const unsigned short* __restrict__ p,
                                              int c0, int c1) {
  union { v16bf v; ushort8 h[2]; } r;
  r.h[0] = *reinterpret_cast<const ushort8*>(p + c0);
  r.h[1] = *reinterpret_cast<const ushort8*>(p + c1);
  return r.v;
}

// PASS 1: accumulate rowSum/colSum of exp(sim)  (no max-shift needed, |sim|<~8)
// PASS 2: recompute sim (identical FP sequence), emit confidence + row/col max
template <int PASS>
__global__ __launch_bounds__(256) void gemm_pass(const unsigned short* __restrict__ x0hi,
                                                 const unsigned short* __restrict__ x0lo,
                                                 const unsigned short* __restrict__ x1hi,
                                                 const unsigned short* __restrict__ x1lo,
                                                 float* __restrict__ rowSum,
                                                 float* __restrict__ colSum,
                                                 unsigned* __restrict__ rowMax,
                                                 unsigned* __restrict__ colMax,
                                                 float* __restrict__ confOut) {
  const int lane = threadIdx.x & 31;
  const int wave = threadIdx.x >> 5;
  const int hh   = lane >> 4;     // half-wave selector (ISA lane-group)
  const int ln   = lane & 15;
  const int n    = blockIdx.z;
  const int lt   = blockIdx.y * 8 + wave;   // 16-row tile index along l
  const int sBase = blockIdx.x * 64;        // 4 consecutive 16-col tiles along s

  const size_t aOff = ((size_t)(n * L_DIM + lt * 16 + ln)) * C_DIM;
  const size_t bOff = ((size_t)(n * S_DIM + sBase + ln)) * C_DIM;
  const unsigned short* aHiRow = x0hi + aOff;
  const unsigned short* aLoRow = x0lo + aOff;
  const unsigned short* bHiRow = x1hi + bOff;
  const unsigned short* bLoRow = x1lo + bOff;

  v8f acc[4] = {};
#pragma unroll
  for (int kb = 0; kb < C_DIM; kb += 32) {
    // A fragment (16x32, MxK): lane ln = M, chunks [kb+8*hh,+8) and [kb+16+8*hh,+8)
    const v16bf aH = load_frag_bf(aHiRow, kb + 8 * hh, kb + 16 + 8 * hh);
    const v16bf aL = load_frag_bf(aLoRow, kb + 8 * hh, kb + 16 + 8 * hh);
#pragma unroll
    for (int t = 0; t < 4; ++t) {
      // B fragment (32x16, KxN): lane ln = N, chunks [kb+8*hh,+8) and [kb+16+8*hh,+8)
      const size_t tOff = (size_t)t * 16 * C_DIM;
      const v16bf bH = load_frag_bf(bHiRow + tOff, kb + 8 * hh, kb + 16 + 8 * hh);
      const v16bf bL = load_frag_bf(bLoRow + tOff, kb + 8 * hh, kb + 16 + 8 * hh);
      // split-bf16 fp32 emulation: hi*hi + hi*lo + lo*hi
      acc[t] = __builtin_amdgcn_wmma_f32_16x16x32_bf16(false, aH, false, bH,
                                                       (short)0, acc[t], false, false);
      acc[t] = __builtin_amdgcn_wmma_f32_16x16x32_bf16(false, aH, false, bL,
                                                       (short)0, acc[t], false, false);
      acc[t] = __builtin_amdgcn_wmma_f32_16x16x32_bf16(false, aL, false, bH,
                                                       (short)0, acc[t], false, false);
    }
  }

  // C/D layout: vgpr j of lane L holds C[M = j + 8*hh][N = ln] of each tile.
  if (PASS == 1) {
    float rowPart[8] = {0.f, 0.f, 0.f, 0.f, 0.f, 0.f, 0.f, 0.f};
#pragma unroll
    for (int t = 0; t < 4; ++t) {
      FVec8 a; a.v = acc[t];
      float colPart = 0.f;
#pragma unroll
      for (int j = 0; j < 8; ++j) {
        const float e = __expf(a.f[j] * SIM_SCALE);
        rowPart[j] += e;
        colPart += e;
      }
      colPart += __shfl_xor(colPart, 16, 32);  // merge M=0..7 with M=8..15 half
      if (lane < 16)
        atomicAdd(&colSum[n * S_DIM + sBase + t * 16 + lane], colPart);
    }
#pragma unroll
    for (int j = 0; j < 8; ++j) {
      float v = rowPart[j];
      v += __shfl_xor(v, 1, 32);
      v += __shfl_xor(v, 2, 32);
      v += __shfl_xor(v, 4, 32);
      v += __shfl_xor(v, 8, 32);
      if (ln == j)
        atomicAdd(&rowSum[n * L_DIM + lt * 16 + 8 * hh + j], v);
    }
  } else {
    float invRow[8];
#pragma unroll
    for (int j = 0; j < 8; ++j)
      invRow[j] = 1.0f / rowSum[n * L_DIM + lt * 16 + 8 * hh + j];
    float rowMaxP[8] = {0.f, 0.f, 0.f, 0.f, 0.f, 0.f, 0.f, 0.f};
#pragma unroll
    for (int t = 0; t < 4; ++t) {
      const float invCol = 1.0f / colSum[n * S_DIM + sBase + t * 16 + ln];
      FVec8 a; a.v = acc[t];
      float colMaxP = 0.f;
#pragma unroll
      for (int j = 0; j < 8; ++j) {
        const float e  = __expf(a.f[j] * SIM_SCALE);
        const float cf = e * e * invRow[j] * invCol;   // bi-softmax confidence
        confOut[((size_t)(n * L_DIM + lt * 16 + 8 * hh + j)) * S_DIM +
                sBase + t * 16 + ln] = cf;
        rowMaxP[j] = fmaxf(rowMaxP[j], cf);
        colMaxP    = fmaxf(colMaxP, cf);
      }
      colMaxP = fmaxf(colMaxP, __shfl_xor(colMaxP, 16, 32));
      if (lane < 16)   // conf >= 0: raw IEEE bits are order-preserving for u32 max
        atomicMax(&colMax[n * S_DIM + sBase + t * 16 + lane], __float_as_uint(colMaxP));
    }
#pragma unroll
    for (int j = 0; j < 8; ++j) {
      float v = rowMaxP[j];
      v = fmaxf(v, __shfl_xor(v, 1, 32));
      v = fmaxf(v, __shfl_xor(v, 2, 32));
      v = fmaxf(v, __shfl_xor(v, 4, 32));
      v = fmaxf(v, __shfl_xor(v, 8, 32));
      if (ln == j)
        atomicMax(&rowMax[n * L_DIM + lt * 16 + 8 * hh + j], __float_as_uint(v));
    }
  }
}

__device__ __forceinline__ bool border_ok(int p) {
  const int h = p >> 6, w = p & (HW - 1);
  return (h >= BORDER_R) && (h < HW - BORDER_R) && (w >= BORDER_R) && (w < HW - BORDER_R);
}

__global__ __launch_bounds__(256) void mask_kernel(const float* __restrict__ conf,
                                                   const unsigned* __restrict__ rowMax,
                                                   const unsigned* __restrict__ colMax,
                                                   float* __restrict__ mask) {
  const size_t total4 = (size_t)N_BATCH * L_DIM * S_DIM / 4;
  const size_t i4 = (size_t)blockIdx.x * blockDim.x + threadIdx.x;
  if (i4 >= total4) return;
  const size_t base = i4 * 4;
  const int s0 = (int)(base & (S_DIM - 1));
  const size_t r = base >> 12;             // / S_DIM
  const int l = (int)(r & (L_DIM - 1));
  const int n = (int)(r >> 12);            // / L_DIM
  const unsigned rm = rowMax[n * L_DIM + l];
  const bool lb = border_ok(l);
  const float4 c = *reinterpret_cast<const float4*>(conf + base);
  const float cv[4] = {c.x, c.y, c.z, c.w};
  float out[4];
#pragma unroll
  for (int e = 0; e < 4; ++e) {
    const int s = s0 + e;
    const unsigned cb = __float_as_uint(cv[e]);
    const bool ok = lb && border_ok(s) && (cv[e] > THRESH) &&
                    (cb == rm) && (cb == colMax[n * S_DIM + s]);
    out[e] = ok ? 1.0f : 0.0f;
  }
  *reinterpret_cast<float4*>(mask + base) = make_float4(out[0], out[1], out[2], out[3]);
}

extern "C" void kernel_launch(void* const* d_in, const int* in_sizes, int n_in,
                              void* d_out, int out_size, void* d_ws, size_t ws_size,
                              hipStream_t stream) {
  (void)in_sizes; (void)n_in; (void)out_size; (void)ws_size;
  const float* x0 = (const float*)d_in[0];
  const float* x1 = (const float*)d_in[1];
  // d_in[2..5] = h0,w0,h1,w1 scalars; fixed to 64 for this problem instance.

  float* out = (float*)d_out;
  const size_t elems = (size_t)N_BATCH * L_DIM * S_DIM;  // 67,108,864
  float* maskOut = out;            // output 0: mask (as 0/1 floats)
  float* confOut = out + elems;    // output 1: confidence

  // Workspace layout: [stats 256KB][x0hi][x0lo][x1hi][x1lo]  (~33.8 MB total)
  char* w = (char*)d_ws;
  float*    rowSum = (float*)w;
  float*    colSum = rowSum + (size_t)N_BATCH * L_DIM;
  unsigned* rowMax = (unsigned*)(colSum + (size_t)N_BATCH * S_DIM);
  unsigned* colMax = rowMax + (size_t)N_BATCH * L_DIM;
  unsigned short* x0hi = (unsigned short*)(w + STATS_BYTES);
  unsigned short* x0lo = x0hi + NE_INPUT;
  unsigned short* x1hi = x0lo + NE_INPUT;
  unsigned short* x1lo = x1hi + NE_INPUT;

  hipMemsetAsync(d_ws, 0, STATS_BYTES, stream);

  const int n4 = (int)(NE_INPUT / 4);                    // 1,048,576
  split_bf16_kernel<<<n4 / 256, 256, 0, stream>>>(x0, x0hi, x0lo, n4);
  split_bf16_kernel<<<n4 / 256, 256, 0, stream>>>(x1, x1hi, x1lo, n4);

  const dim3 grid(S_DIM / 64, L_DIM / (16 * 8), N_BATCH);  // (64, 32, 4)
  gemm_pass<1><<<grid, 256, 0, stream>>>(x0hi, x0lo, x1hi, x1lo,
                                         rowSum, colSum, rowMax, colMax, confOut);
  gemm_pass<2><<<grid, 256, 0, stream>>>(x0hi, x0lo, x1hi, x1lo,
                                         rowSum, colSum, rowMax, colMax, confOut);

  const size_t total4 = elems / 4;
  mask_kernel<<<(unsigned)((total4 + 255) / 256), 256, 0, stream>>>(confOut, rowMax, colMax, maskOut);
}
